// AttentionBlockHo_12000138625654
// MI455X (gfx1250) — compile-verified, matrix operand order checked by reading, not run on hardware
//
#include <hip/hip_runtime.h>
#include <stdint.h>

// ---------------------------------------------------------------------------
// AttentionBlock (GroupNorm -> QKV -> softmax(QK^T)V -> proj + residual)
// CDNA5 / gfx1250: wave32, bf16 WMMA (V_WMMA_F32_16X16X32_BF16),
// TDM (tensor_load_to_lds) block-level staging, flash-style attention.
// ---------------------------------------------------------------------------

typedef __attribute__((ext_vector_type(16))) __bf16 v16bf;
typedef __attribute__((ext_vector_type(8)))  float  v8f;
typedef __attribute__((ext_vector_type(4)))  unsigned int v4u;
typedef __attribute__((ext_vector_type(8)))  int v8i;
typedef __attribute__((ext_vector_type(4)))  int v4i;

#define BATCH 8
#define CHN   256
#define NSP   4096
#define NGRP  32
#define CPG   8      // channels per group
#define HEADS 4
#define CHD   64     // channels per head
#define EPSV  1e-5f

#if __has_builtin(__builtin_amdgcn_tensor_load_to_lds) && \
    __has_builtin(__builtin_amdgcn_s_wait_tensorcnt)
#define USE_TDM 1
#endif

union FragU { v16bf v; uint32_t u[8]; };

__device__ __forceinline__ unsigned short f2bf(float x) {
  union { float f; uint32_t u; } c; c.f = x;
  uint32_t u = c.u;
  u += 0x7fffu + ((u >> 16) & 1u);   // round-to-nearest-even
  return (unsigned short)(u >> 16);
}

// Load a 16-bit A/B WMMA fragment where K runs contiguously from `base`.
// ISA layout (16-bit 16x32): lanes 0-15 hold K = {0..7, 16..23},
// lanes 16-31 hold K = {8..15, 24..31}; two elements per VGPR.
__device__ __forceinline__ v16bf load_frag(const unsigned short* base, int half) {
  const uint32_t* p = (const uint32_t*)base;
  FragU f;
  const int o0 = half * 4;       // dword offset of K = half*8
  const int o1 = 8 + half * 4;   // dword offset of K = 16 + half*8
  f.u[0] = p[o0 + 0]; f.u[1] = p[o0 + 1]; f.u[2] = p[o0 + 2]; f.u[3] = p[o0 + 3];
  f.u[4] = p[o1 + 0]; f.u[5] = p[o1 + 1]; f.u[6] = p[o1 + 2]; f.u[7] = p[o1 + 3];
  return f.v;
}

__device__ __forceinline__ v8f wmma_bf16(v16bf a, v16bf b, v8f c) {
  return __builtin_amdgcn_wmma_f32_16x16x32_bf16(
      /*neg_a=*/false, a, /*neg_b=*/false, b,
      /*c_mod=*/(short)0, c, /*reuse_a=*/false, /*reuse_b=*/false);
}

#ifdef USE_TDM
// Issue one 2D TDM tile load (global -> LDS), bf16 elements (data_size=1).
// D# per CDNA5 ISA ch.8: group0 = {count=1, lds_addr, global_addr, type=2},
// group1 = {data_size, tensor_dim0/1, tile_dim0/1, tensor_dim0_stride}.
// This toolchain exposes the 6-arg builtin:
//   (uint32x4 g0, int32x8 g1, int32x4 g2, int32x4 g3, int32x8 g4, i32 cpol)
__device__ __forceinline__ void tdm_load_2d(void* lds_dst, const void* gsrc,
                                            uint32_t tile_d0, uint32_t tile_d1,
                                            uint32_t tensor_d0, uint32_t tensor_d1,
                                            uint32_t stride0) {
  uint64_t ga = (uint64_t)(uintptr_t)gsrc;
  uint32_t lo = (uint32_t)(uintptr_t)lds_dst;  // LDS byte offset (low 32 bits)
  v4u g0;
  g0[0] = 1u;                                            // count=1, user D#
  g0[1] = lo;                                            // lds_addr
  g0[2] = (uint32_t)ga;                                  // global_addr[31:0]
  g0[3] = (uint32_t)((ga >> 32) & 0x01FFFFFFu) | (2u << 30);  // [56:32] | type=2
  v8i g1;
  g1[0] = (int)(1u << 16);                               // data_size=1 (2 bytes)
  g1[1] = (int)((tensor_d0 & 0xFFFFu) << 16);            // tensor_dim0 lo16
  g1[2] = (int)(((tensor_d0 >> 16) & 0xFFFFu) | ((tensor_d1 & 0xFFFFu) << 16));
  g1[3] = (int)(((tensor_d1 >> 16) & 0xFFFFu) | ((tile_d0 & 0xFFFFu) << 16));
  g1[4] = (int)(tile_d1 & 0xFFFFu);                      // tile_dim1, tile_dim2=0
  g1[5] = (int)stride0;                                  // dim0 stride [31:0]
  g1[6] = 0;
  g1[7] = 0;
  v4i z4 = {0, 0, 0, 0};
  v8i z8 = {0, 0, 0, 0, 0, 0, 0, 0};
  __builtin_amdgcn_tensor_load_to_lds(g0, g1, z4, z4, z8, 0);
}
#endif

// ---------------------------------------------------------------------------
// 1) GroupNorm statistics: one block per (b, g); reduce 8*4096 elements.
// ---------------------------------------------------------------------------
__global__ void __launch_bounds__(256)
gn_stats_kernel(const float* __restrict__ x, float* __restrict__ stats) {
  __shared__ float sh_s[256];
  __shared__ float sh_q[256];
  const int bg = blockIdx.x;                       // b*32 + g
  const float* base = x + (size_t)bg * (CPG * NSP);
  float s = 0.f, q = 0.f;
  for (int i = threadIdx.x; i < CPG * NSP; i += 256) {
    float v = base[i];
    s += v; q += v * v;
  }
  sh_s[threadIdx.x] = s; sh_q[threadIdx.x] = q;
  __syncthreads();
  for (int off = 128; off > 0; off >>= 1) {
    if (threadIdx.x < off) {
      sh_s[threadIdx.x] += sh_s[threadIdx.x + off];
      sh_q[threadIdx.x] += sh_q[threadIdx.x + off];
    }
    __syncthreads();
  }
  if (threadIdx.x == 0) {
    const float inv = 1.0f / (float)(CPG * NSP);
    float mean = sh_s[0] * inv;
    float var  = sh_q[0] * inv - mean * mean;
    stats[bg * 2 + 0] = mean;
    stats[bg * 2 + 1] = rsqrtf(var + EPSV);
  }
}

// ---------------------------------------------------------------------------
// 2) Apply GroupNorm, convert to bf16, transpose to (b, n, c).
// ---------------------------------------------------------------------------
__global__ void __launch_bounds__(256)
gn_apply_kernel(const float* __restrict__ x, const float* __restrict__ stats,
                const float* __restrict__ gw, const float* __restrict__ gb,
                unsigned short* __restrict__ xn) {
  size_t idx = (size_t)blockIdx.x * 256 + threadIdx.x;   // over B*C*N
  int n = (int)(idx % NSP);
  int c = (int)((idx / NSP) % CHN);
  int b = (int)(idx / ((size_t)NSP * CHN));
  int g = c / CPG;
  float mean = stats[(b * NGRP + g) * 2 + 0];
  float rstd = stats[(b * NGRP + g) * 2 + 1];
  float v = (x[idx] - mean) * rstd * gw[c] + gb[c];
  xn[((size_t)b * NSP + n) * CHN + c] = f2bf(v);
}

// ---------------------------------------------------------------------------
// 3) fp32 -> bf16 weight conversion.
// ---------------------------------------------------------------------------
__global__ void __launch_bounds__(256)
f2bf_kernel(const float* __restrict__ src, unsigned short* __restrict__ dst, int n) {
  int i = blockIdx.x * 256 + threadIdx.x;
  if (i < n) dst[i] = f2bf(src[i]);
}

// ---------------------------------------------------------------------------
// 4) QKV GEMM (768x256)x(256x4096) per batch. Block = 128 threads (4 waves),
//    block tile 64(M)x64(N), full-K A/B panels TDM-staged into LDS once.
// ---------------------------------------------------------------------------
__global__ void __launch_bounds__(128)
qkv_gemm_kernel(const unsigned short* __restrict__ wqkv,
                const unsigned short* __restrict__ xn,
                const float* __restrict__ bias,
                unsigned short* __restrict__ qo,
                unsigned short* __restrict__ ko,
                unsigned short* __restrict__ vo) {
  __shared__ unsigned short Asm[64 * CHN];   // 32 KB: 64 M-rows x 256 K
  __shared__ unsigned short Bsm[64 * CHN];   // 32 KB: 64 N-cols x 256 K
  const int lane = threadIdx.x & 31;
  const int wave = threadIdx.x >> 5;
  const int half = lane >> 4;
  const int lid  = lane & 15;
  const int mbase = blockIdx.x * 64;        // 0..704
  const int n0    = blockIdx.y * 64;
  const int b     = blockIdx.z;

  const unsigned short* ag = wqkv + (size_t)mbase * CHN;
  const unsigned short* bg = xn + ((size_t)b * NSP + n0) * CHN;

#ifdef USE_TDM
  if (wave == 0) {
    tdm_load_2d(Asm, ag, CHN, 64, CHN, 64, CHN);
    tdm_load_2d(Bsm, bg, CHN, 64, CHN, 64, CHN);
    __builtin_amdgcn_s_wait_tensorcnt(0);
  }
#else
  {
    const uint4* as = (const uint4*)ag; uint4* ad = (uint4*)Asm;
    const uint4* bs = (const uint4*)bg; uint4* bd = (uint4*)Bsm;
    for (int t = threadIdx.x; t < 64 * CHN / 8; t += 128) { ad[t] = as[t]; bd[t] = bs[t]; }
  }
#endif
  __syncthreads();

  v8f acc[4] = {};
  for (int kb = 0; kb < CHN; kb += 32) {
    v16bf a = load_frag(Asm + (size_t)(wave * 16 + lid) * CHN + kb, half);
#pragma unroll
    for (int j = 0; j < 4; ++j) {
      v16bf bf = load_frag(Bsm + (size_t)(j * 16 + lid) * CHN + kb, half);
      acc[j] = wmma_bf16(a, bf, acc[j]);
    }
  }

  const float scale = 0.35355339059327373f;  // 64^-0.25, applied to q and k
#pragma unroll
  for (int j = 0; j < 4; ++j) {
#pragma unroll
    for (int r = 0; r < 8; ++r) {
      int o = mbase + wave * 16 + r + half * 8;  // D layout: row = r + half*8
      int n = n0 + j * 16 + lid;                 //           col = lane % 16
      float val = acc[j][r] + bias[o];
      if (o < CHN) {
        int head = o >> 6, c = o & 63;
        qo[(((size_t)(b * HEADS + head) * NSP) + n) * CHD + c] = f2bf(val * scale);
      } else if (o < 2 * CHN) {
        int oc = o - CHN, head = oc >> 6, c = oc & 63;
        ko[(((size_t)(b * HEADS + head) * NSP) + n) * CHD + c] = f2bf(val * scale);
      } else {
        int oc = o - 2 * CHN, head = oc >> 6, c = oc & 63;
        vo[(((size_t)(b * HEADS + head) * CHD) + c) * NSP + n] = f2bf(val);
      }
    }
  }
}

// ---------------------------------------------------------------------------
// 5) Flash attention: block = 8 waves sharing one (b,head); each wave owns 16
//    queries. K/V tiles (32 keys) are TDM double-buffered through LDS; per
//    step: 4 WMMA (scores) + online softmax + 4 WMMA (P*V).
// ---------------------------------------------------------------------------
__global__ void __launch_bounds__(256)
flash_attn_kernel(const unsigned short* __restrict__ q,
                  const unsigned short* __restrict__ k,
                  const unsigned short* __restrict__ v,
                  unsigned short* __restrict__ aout) {
  __shared__ unsigned short Kb[2][32 * CHD];     // 2 x 4 KB (32 keys x 64 ch)
  __shared__ unsigned short Vb[2][CHD * 32];     // 2 x 4 KB (64 ch x 32 keys)
  __shared__ unsigned short Pb[8 * 16 * 32];     // 8 KB P scratch (per wave)
  const int lane = threadIdx.x & 31;
  const int wave = threadIdx.x >> 5;
  const int half = lane >> 4;
  const int lid  = lane & 15;
  const int bh = blockIdx.x;                     // b*4 + head
  const int qt = blockIdx.y * 8 + wave;          // 16-query tile, 0..255
  unsigned short* pw = &Pb[wave * 512];

  const unsigned short* kg = k + (size_t)bh * NSP * CHD;  // (n, 64) rows dense
  const unsigned short* vg = v + (size_t)bh * CHD * NSP;  // (64, n) row stride N

#ifdef USE_TDM
  auto issue_kv = [&](int buf, int s0) {
    // K tile: 32 rows(keys) x 64 ch, row stride 64 (dense)
    tdm_load_2d(&Kb[buf][0], kg + (size_t)s0 * CHD, CHD, 32, CHD, NSP, CHD);
    // V tile: 64 rows(ch) x 32 keys, row stride 4096
    tdm_load_2d(&Vb[buf][0], vg + s0, 32, CHD, NSP, CHD, NSP);
  };
  if (wave == 0) { issue_kv(0, 0); __builtin_amdgcn_s_wait_tensorcnt(0); }
  __syncthreads();
#endif

  const unsigned short* qbase = q + ((size_t)bh * NSP + qt * 16 + lid) * CHD;
  v16bf qa0 = load_frag(qbase,      half);       // K = c 0..31
  v16bf qa1 = load_frag(qbase + 32, half);       // K = c 32..63

  float m[8], l[8];
  v8f o[4] = {};
#pragma unroll
  for (int r = 0; r < 8; ++r) { m[r] = -1e30f; l[r] = 0.f; }

  const int steps = NSP / 32;
  for (int it = 0; it < steps; ++it) {
    const int s0 = it * 32;
#ifdef USE_TDM
    const int cur = it & 1;
    if (wave == 0 && it + 1 < steps) issue_kv(cur ^ 1, s0 + 32);
#else
    const int cur = 0;
    {
      // cooperative stage: K tile is 4 KB contiguous, V tile 64 rows x 64 B
      const uint4* ks = (const uint4*)(kg + (size_t)s0 * CHD);
      ((uint4*)&Kb[0][0])[threadIdx.x] = ks[threadIdx.x];
      int row = threadIdx.x >> 2, ch4 = threadIdx.x & 3;
      ((uint4*)&Vb[0][0])[threadIdx.x] =
          *(const uint4*)(vg + (size_t)row * NSP + s0 + ch4 * 8);
    }
    __syncthreads();
#endif
    v8f s01[2] = {};
#pragma unroll
    for (int sub = 0; sub < 2; ++sub) {
      const unsigned short* kcol = &Kb[cur][(sub * 16 + lid) * CHD];
      v16bf kb0 = load_frag(kcol,      half);
      v16bf kb1 = load_frag(kcol + 32, half);
      s01[sub] = wmma_bf16(qa0, kb0, s01[sub]);
      s01[sub] = wmma_bf16(qa1, kb1, s01[sub]);
    }
    // Online softmax; D layout: VGPR r -> row r + half*8, lane%16 -> column.
#pragma unroll
    for (int r = 0; r < 8; ++r) {
      float cm = fmaxf(s01[0][r], s01[1][r]);
      cm = fmaxf(cm, __shfl_xor(cm, 8));
      cm = fmaxf(cm, __shfl_xor(cm, 4));
      cm = fmaxf(cm, __shfl_xor(cm, 2));
      cm = fmaxf(cm, __shfl_xor(cm, 1));
      float mnew = fmaxf(m[r], cm);
      float p0 = __expf(s01[0][r] - mnew);
      float p1 = __expf(s01[1][r] - mnew);
      float rs = p0 + p1;
      rs += __shfl_xor(rs, 8);
      rs += __shfl_xor(rs, 4);
      rs += __shfl_xor(rs, 2);
      rs += __shfl_xor(rs, 1);
      float alpha = __expf(m[r] - mnew);
      l[r] = l[r] * alpha + rs;
      m[r] = mnew;
#pragma unroll
      for (int j = 0; j < 4; ++j) o[j][r] *= alpha;
      pw[(r + half * 8) * 32 + lid]      = f2bf(p0);
      pw[(r + half * 8) * 32 + 16 + lid] = f2bf(p1);
    }
    __syncthreads();
    v16bf pa = load_frag(pw + lid * 32, half);   // A fragment: row=lid, K=s
#pragma unroll
    for (int j = 0; j < 4; ++j) {
      v16bf vb = load_frag(&Vb[cur][(j * 16 + lid) * 32], half);  // B: col=c, K=s
      o[j] = wmma_bf16(pa, vb, o[j]);
    }
#ifdef USE_TDM
    if (wave == 0) __builtin_amdgcn_s_wait_tensorcnt(0);
#endif
    __syncthreads();
  }

  const int b = bh >> 2, head = bh & 3;
#pragma unroll
  for (int r = 0; r < 8; ++r) {
    float inv = 1.0f / l[r];
    int n = qt * 16 + r + half * 8;
#pragma unroll
    for (int j = 0; j < 4; ++j) {
      int c = head * CHD + j * 16 + lid;
      aout[((size_t)b * NSP + n) * CHN + c] = f2bf(o[j][r] * inv);
    }
  }
}

// ---------------------------------------------------------------------------
// 6) proj GEMM (256x256)x(256x4096) + bias + residual -> fp32 d_out (b,c,n).
//    Same TDM-staged block tiling as the QKV GEMM.
// ---------------------------------------------------------------------------
__global__ void __launch_bounds__(128)
proj_gemm_kernel(const unsigned short* __restrict__ wp,
                 const unsigned short* __restrict__ a,
                 const float* __restrict__ bias,
                 const float* __restrict__ x,
                 float* __restrict__ out) {
  __shared__ unsigned short Asm[64 * CHN];
  __shared__ unsigned short Bsm[64 * CHN];
  const int lane = threadIdx.x & 31;
  const int wave = threadIdx.x >> 5;
  const int half = lane >> 4;
  const int lid  = lane & 15;
  const int mbase = blockIdx.x * 64;        // 0..192
  const int n0    = blockIdx.y * 64;
  const int b     = blockIdx.z;

  const unsigned short* ag = wp + (size_t)mbase * CHN;
  const unsigned short* bg = a + ((size_t)b * NSP + n0) * CHN;

#ifdef USE_TDM
  if (wave == 0) {
    tdm_load_2d(Asm, ag, CHN, 64, CHN, 64, CHN);
    tdm_load_2d(Bsm, bg, CHN, 64, CHN, 64, CHN);
    __builtin_amdgcn_s_wait_tensorcnt(0);
  }
#else
  {
    const uint4* as = (const uint4*)ag; uint4* ad = (uint4*)Asm;
    const uint4* bs = (const uint4*)bg; uint4* bd = (uint4*)Bsm;
    for (int t = threadIdx.x; t < 64 * CHN / 8; t += 128) { ad[t] = as[t]; bd[t] = bs[t]; }
  }
#endif
  __syncthreads();

  v8f acc[4] = {};
  for (int kb = 0; kb < CHN; kb += 32) {
    v16bf af = load_frag(Asm + (size_t)(wave * 16 + lid) * CHN + kb, half);
#pragma unroll
    for (int j = 0; j < 4; ++j) {
      v16bf bf = load_frag(Bsm + (size_t)(j * 16 + lid) * CHN + kb, half);
      acc[j] = wmma_bf16(af, bf, acc[j]);
    }
  }
#pragma unroll
  for (int j = 0; j < 4; ++j) {
#pragma unroll
    for (int r = 0; r < 8; ++r) {
      int o = mbase + wave * 16 + r + half * 8;
      int n = n0 + j * 16 + lid;
      size_t idx = ((size_t)b * CHN + o) * NSP + n;
      out[idx] = x[idx] + acc[j][r] + bias[o];
    }
  }
}

// ---------------------------------------------------------------------------
// Workspace layout (bytes)
// ---------------------------------------------------------------------------
#define OFF_STATS 0                                          // 256*2*4 = 2 KB
#define OFF_WQ    ((size_t)4096)                             // 768*256 bf16
#define OFF_WP    (OFF_WQ + (size_t)3 * CHN * CHN * 2)
#define OFF_XN    (OFF_WP + (size_t)CHN * CHN * 2)           // (b,n,c) bf16
#define OFF_Q     (OFF_XN + (size_t)BATCH * NSP * CHN * 2)   // (bh,n,64) bf16
#define OFF_K     (OFF_Q  + (size_t)BATCH * CHN * NSP * 2)   // (bh,n,64) bf16
#define OFF_V     (OFF_K  + (size_t)BATCH * CHN * NSP * 2)   // (bh,64,n) bf16
#define OFF_A     (OFF_V  + (size_t)BATCH * CHN * NSP * 2)   // (b,n,c) bf16

extern "C" void kernel_launch(void* const* d_in, const int* in_sizes, int n_in,
                              void* d_out, int out_size, void* d_ws, size_t ws_size,
                              hipStream_t stream) {
  const float* x      = (const float*)d_in[0];
  const float* norm_w = (const float*)d_in[1];
  const float* norm_b = (const float*)d_in[2];
  const float* qkv_w  = (const float*)d_in[3];
  const float* qkv_b  = (const float*)d_in[4];
  const float* proj_w = (const float*)d_in[5];
  const float* proj_b = (const float*)d_in[6];

  char* ws = (char*)d_ws;
  float*          stats = (float*)(ws + OFF_STATS);
  unsigned short* wq    = (unsigned short*)(ws + OFF_WQ);
  unsigned short* wp    = (unsigned short*)(ws + OFF_WP);
  unsigned short* xn    = (unsigned short*)(ws + OFF_XN);
  unsigned short* qb    = (unsigned short*)(ws + OFF_Q);
  unsigned short* kb    = (unsigned short*)(ws + OFF_K);
  unsigned short* vb    = (unsigned short*)(ws + OFF_V);
  unsigned short* ab    = (unsigned short*)(ws + OFF_A);

  gn_stats_kernel<<<BATCH * NGRP, 256, 0, stream>>>(x, stats);
  gn_apply_kernel<<<(BATCH * CHN * NSP) / 256, 256, 0, stream>>>(
      x, stats, norm_w, norm_b, xn);
  f2bf_kernel<<<(3 * CHN * CHN + 255) / 256, 256, 0, stream>>>(qkv_w, wq, 3 * CHN * CHN);
  f2bf_kernel<<<(CHN * CHN + 255) / 256, 256, 0, stream>>>(proj_w, wp, CHN * CHN);

  qkv_gemm_kernel<<<dim3(12, 64, BATCH), 128, 0, stream>>>(wq, xn, qkv_b, qb, kb, vb);
  flash_attn_kernel<<<dim3(BATCH * HEADS, NSP / (16 * 8)), 256, 0, stream>>>(qb, kb, vb, ab);
  proj_gemm_kernel<<<dim3(4, 64, BATCH), 128, 0, stream>>>(wp, ab, proj_b, x, (float*)d_out);
}
